// SmartSplat_30751965839963
// MI455X (gfx1250) — matrix operand order repaired and checked by reading, not compiled, not run on hardware
//
#include <hip/hip_runtime.h>
#include <hip/hip_bf16.h>
#include <math.h>

typedef __attribute__((ext_vector_type(16))) _Float16 v16h;
typedef __attribute__((ext_vector_type(8)))  float    v8f;
typedef __attribute__((ext_vector_type(4)))  unsigned int u32x4;
typedef __attribute__((ext_vector_type(8)))  int          i32x8;
typedef __attribute__((ext_vector_type(4)))  int          i32x4;

#define HH 512
#define WW 512
#define NGAUSS 2048
#define NCHUNK (NGAUSS / 32)            // 64 K-chunks of 32 gaussians
#define TWO_PI_F 6.28318530717958647692f
#define LOG2E_F  1.44269504088896340736f
#define A_MIN (1.0f / 255.0f)
#define A_MAX 0.999f

// ---------------------------------------------------------------------------
// Kernel 1: per-gaussian bake (N=2048, trivial).
// P4 = (mean_x, a2, b2, mean_y) with a2 = -log2e*0.5*c0, b2 = -log2e*c1
// Q2 = (c2q, l2op)             with c2q = -log2e*0.5*c2, l2op = log2(op)
// so that  log2(alpha) = a2*dx^2 + (b2*dy)*dx + (c2q*dy^2 + l2op)
// ---------------------------------------------------------------------------
__global__ void prep_params(const float* __restrict__ xyz,
                            const float* __restrict__ scaling,
                            const float* __restrict__ rotation,
                            const float* __restrict__ opacity,
                            float4* __restrict__ p4,
                            float2* __restrict__ q2, int n) {
    int g = blockIdx.x * blockDim.x + threadIdx.x;
    if (g >= n) return;
    float r     = rotation[g];
    float theta = TWO_PI_F / (1.0f + __expf(-r));     // sigmoid * 2pi
    float sx = fabsf(scaling[2 * g + 0]);
    float sy = fabsf(scaling[2 * g + 1]);
    float cs = __cosf(theta), sn = __sinf(theta);
    float s0 = sx * sx, s1 = sy * sy;
    float A = cs * cs * s0 + sn * sn * s1;
    float B = cs * sn * (s0 - s1);
    float C = sn * sn * s0 + cs * cs * s1;
    float inv_det = 1.0f / (A * C - B * B);
    float c0 = C * inv_det, c1 = -B * inv_det, c2 = A * inv_det;
    float mx = 0.5f * ((xyz[2 * g + 0] + 1.0f) * (float)WW - 1.0f);
    float my = 0.5f * ((xyz[2 * g + 1] + 1.0f) * (float)HH - 1.0f);
    float op = opacity[g];
    float a2 = -LOG2E_F * 0.5f * c0;
    float b2 = -LOG2E_F * c1;
    float c2q = -LOG2E_F * 0.5f * c2;
    float l2op = (op > 0.0f) ? log2f(op) : -1.0e30f;
    p4[g] = make_float4(mx, a2, b2, my);
    q2[g] = make_float2(c2q, l2op);
}

// ---------------------------------------------------------------------------
// Kernel 2: swizzle features (N x 3) into the WMMA B-fragment order for
// V_WMMA_F32_16X16X32_F16: lane holds column N=lane&15 (channel; 3 used),
// K = (lane&16 ? 8 : 0) + (e&7) + (e&8 ? 16 : 0). Flat index = ((chunk*32 + lane)*16 + e).
// ---------------------------------------------------------------------------
__global__ void prep_bmat(const float* __restrict__ features,
                          _Float16* __restrict__ bbuf, int n) {
    int idx = blockIdx.x * blockDim.x + threadIdx.x;
    if (idx >= NCHUNK * 32 * 16) return;
    int e    = idx & 15;
    int lane = (idx >> 4) & 31;
    int ch   = idx >> 9;
    int c    = lane & 15;
    int kb   = (lane & 16) ? 8 : 0;
    int g    = ch * 32 + kb + (e & 7) + ((e & 8) ? 16 : 0);
    float v  = 0.0f;
    if (c < 3 && g < n) v = features[g * 3 + c];
    bbuf[idx] = (_Float16)v;
}

// ---------------------------------------------------------------------------
// Kernel 3: main splat. One wave32 per 16-pixel x-strip; 8 waves/block share
// one image row. Raw params are DMA'd into LDS with the Tensor Data Mover,
// then specialized per-row in place:  sP[g] = (mx, a2, B2=b2*dy, C2).
// Inner loop per alpha: 1 ds_load_b128 + pk_sub + 2x pk_fma + raw v_exp_f32
// (+ min/cmp/sel + cvt_pk) feeding v_wmma_f32_16x16x32_f16.
// ---------------------------------------------------------------------------
__global__ void __launch_bounds__(256) splat_wmma(
        const float4* __restrict__ gP4, const float2* __restrict__ gQ2,
        const _Float16* __restrict__ bbuf, float* __restrict__ out) {
    __shared__ float4 sP[NGAUSS];    // 32 KB
    int tid  = threadIdx.x;
    int lane = tid & 31;
    int wave = tid >> 5;

    // --- TDM: DMA 2048 * 16B of raw params into LDS (wave 0 only) ---------
    if (wave == 0) {
        unsigned long long ga = (unsigned long long)(uintptr_t)gP4;
        unsigned int lds_addr = (unsigned int)(uintptr_t)&sP[0]; // low 32 = LDS offset
        u32x4 g0;
        g0.x = 1u;                                        // count=1, user mode
        g0.y = lds_addr;                                  // LDS dest (bytes)
        g0.z = (unsigned int)(ga & 0xFFFFFFFFu);          // global_addr[31:0]
        g0.w = (unsigned int)((ga >> 32) & 0x01FFFFFFu)   // global_addr[56:32]
             | (2u << 30);                                // type = 2 (image)
        i32x8 g1;
        g1[0] = 0x00020000;    // wg_mask=0, data_size=2 (4B), no flags
        g1[1] = 0x20000000;    // tensor_dim0[15:0]=8192 in bits[31:16]
        g1[2] = 0x00010000;    // tensor_dim0 hi=0, tensor_dim1=1
        g1[3] = 0x20000000;    // tile_dim0 = 8192 dwords (32 KB)
        g1[4] = 1;             // tile_dim1 = 1, tile_dim2 = 0
        g1[5] = 0x00002000;    // tensor_dim0_stride = 8192
        g1[6] = 0;
        g1[7] = 0;
        i32x4 z4 = {};
        i32x8 z8 = {};
        __builtin_amdgcn_tensor_load_to_lds(g0, g1, z4, z4, z8, 0);
        __builtin_amdgcn_s_wait_tensorcnt(0);
    }
    __syncthreads();

    // --- per-row specialization in place ----------------------------------
    float rowy = (float)(blockIdx.x >> 2);   // all 8 waves share this row
    for (int i = tid; i < NGAUSS; i += 256) {
        float4 p = sP[i];                    // (mx, a2, b2, my)
        float2 q = gQ2[i];                   // (c2q, l2op)
        float dy = rowy - p.w;
        sP[i] = make_float4(p.x, p.y, p.z * dy, fmaf(q.x * dy, dy, q.y));
    }
    __syncthreads();

    int tile = blockIdx.x * 8 + wave;        // 16384 strips total
    int ty   = tile >> 5;
    int tx0  = (tile & 31) << 4;
    int hi8  = (lane & 16) ? 8 : 0;
    float px = (float)(tx0 + (lane & 15));   // A-matrix row M = lane&15

    const v16h* __restrict__ Bfrag = (const v16h*)bbuf;
    v8f acc = {};

    for (int ch = 0; ch < NCHUNK; ++ch) {
        if (ch + 1 < NCHUNK)
            __builtin_prefetch(&Bfrag[(ch + 1) * 32 + lane], 0, 1); // global_prefetch_b8
        v16h b = Bfrag[ch * 32 + lane];
        int gbase = ch * 32;
        v16h a;
#pragma unroll
        for (int e = 0; e < 8; ++e) {
            // A-fragment K map: VGPRs 0-3 -> K = hi8+e, VGPRs 4-7 -> K = 16+hi8+e
            int g1 = gbase + hi8 + e;
            int g2 = g1 + 16;
            float4 p = sP[g1];
            float dx = px - p.x;
            float t  = fmaf(fmaf(p.y, dx, p.z), dx, p.w);
            // raw v_exp_f32: alpha < 2^-126 is far below the 1/255 cutoff,
            // so denorm-safe exp2 range fixup is pointless here.
            float al = fminf(__builtin_amdgcn_exp2f(t), A_MAX);
            a[e] = (_Float16)((al < A_MIN) ? 0.0f : al);

            p  = sP[g2];
            dx = px - p.x;
            t  = fmaf(fmaf(p.y, dx, p.z), dx, p.w);
            al = fminf(__builtin_amdgcn_exp2f(t), A_MAX);
            a[e + 8] = (_Float16)((al < A_MIN) ? 0.0f : al);
        }
        // D = A(alpha 16x32 f16) * B(features 32x16 f16) + C, f32 accum
        acc = __builtin_amdgcn_wmma_f32_16x16x32_f16(
            false, a, false, b, (short)0, acc, false, false);
    }

    // C/D layout: lane holds N = lane&15 (channel), VGPR r -> M = r + hi8.
    int c = lane & 15;
    if (c < 3) {
        float* o = out + c * (HH * WW) + ty * WW + tx0 + hi8;
#pragma unroll
        for (int r = 0; r < 8; ++r) {
            float v = acc[r];
            o[r] = fminf(fmaxf(v, 0.0f), 1.0f);
        }
    }
}

// ---------------------------------------------------------------------------
extern "C" void kernel_launch(void* const* d_in, const int* in_sizes, int n_in,
                              void* d_out, int out_size, void* d_ws, size_t ws_size,
                              hipStream_t stream) {
    const float* xyz      = (const float*)d_in[0];
    const float* scaling  = (const float*)d_in[1];
    const float* rotation = (const float*)d_in[2];
    const float* features = (const float*)d_in[3];
    const float* opacity  = (const float*)d_in[4];
    float* out = (float*)d_out;

    char* ws = (char*)d_ws;
    float4*   p4   = (float4*)(ws);                          // 32 KB
    float2*   q2   = (float2*)(ws + (size_t)NGAUSS * 16);    // 16 KB
    _Float16* bbuf = (_Float16*)(ws + (size_t)NGAUSS * 24);  // 64 KB

    int n = in_sizes[2];  // rotation is (N,1) -> N = 2048

    prep_params<<<(n + 255) / 256, 256, 0, stream>>>(xyz, scaling, rotation,
                                                     opacity, p4, q2, n);
    int belems = NCHUNK * 32 * 16;
    prep_bmat<<<(belems + 255) / 256, 256, 0, stream>>>(features, bbuf, n);

    // 16384 strips (512 rows x 32 strips), 8 waves per block, 1 row per block
    splat_wmma<<<2048, 256, 0, stream>>>(p4, q2, bbuf, out);
}
